// SpatialDGCNN_50714973831363
// MI455X (gfx1250) — matrix operand order repaired and checked by reading, not compiled, not run on hardware
//
#include <hip/hip_runtime.h>
#include <hip/hip_bf16.h>
#include <math.h>

// ---------------- problem constants (match reference) ----------------
#define BB   2
#define NN   2048
#define KK   16      // neighbors
#define HH   4       // heads
#define CC   8       // classes
#define PPAIR 36     // C*(C+1)/2
#define PE_DIM 64
#define EMBD 1024
#define NEGS 0.2f
#define COLS  (BB*NN*KK)   // 65536 gemm columns for per-neighbor tensors
#define PCOLS (BB*NN)      // 4096 per-point columns
#define XCF   2048         // concat feature width (256+256+512+1024)

typedef __attribute__((ext_vector_type(16))) _Float16 v16h;
typedef __attribute__((ext_vector_type(8)))  float    v8f;

__device__ __forceinline__ float lrelu_(float x){ return x > 0.f ? x : NEGS * x; }

// invs = 1/sqrt(1+1e-5)
#define INVS 0.9999950000374997f

// ---------------- ball query: per point, first K in-ball indices ----------------
__global__ void k_ball(const float* __restrict__ pos, int* __restrict__ idx){
  int t = blockIdx.x * blockDim.x + threadIdx.x;
  if (t >= BB * NN) return;
  int b = t / NN, n = t % NN;
  const float* pb = pos + (size_t)b * NN * 2;
  float x = pb[n * 2], y = pb[n * 2 + 1];
  float sqn = x * x + y * y;
  const float r2 = 0.08f * 0.08f;
  int out[KK]; int cnt = 0;
  for (int m = 0; m < NN; ++m){
    float mx = pb[m * 2], my = pb[m * 2 + 1];
    float sqm = mx * mx + my * my;
    float d = sqn + sqm - 2.f * (x * mx + y * my);
    if (!(d > r2)){            // keep: d <= r^2 (same formula as reference)
      out[cnt++] = m;
      if (cnt == KK) break;
    }
  }
  int f = (cnt > 0) ? out[0] : n;
  for (int k = cnt; k < KK; ++k) out[k] = f;
  for (int k = 0; k < KK; ++k) idx[(size_t)t * KK + k] = out[k];
}

// ---------------- pack f32 weights into WMMA A-fragment-ready f16 tiles --------
// A fragment (16-bit A 16x32), contiguous k order matching B loads:
//  lanes 0-15 : element e -> K=e      (M = lane)
//  lanes 16-31: element e -> K=16+e   (M = lane-16)
// packed[( (mt*nKt+kt)*32 + lane )*16 + e]
__global__ void k_pack(const float* __restrict__ W, _Float16* __restrict__ out,
                       int M, int Kd, int F, int transHeads){
  long long total = (long long)M * Kd;
  long long p = (long long)blockIdx.x * blockDim.x + threadIdx.x;
  if (p >= total) return;
  int nKt    = Kd >> 5;
  int tIdx   = (int)(p >> 9);       // 512 packed elems per 16x32 tile
  int within = (int)(p & 511);
  int lane = within >> 4, e = within & 15;
  int mt = tIdx / nKt, kt = tIdx % nKt;
  int halfw = lane >> 4;
  int m = mt * 16 + (lane & 15);
  int j = e >> 1, pbit = e & 1;
  int base = (j < 4) ? (halfw ? 8 : 0) : (halfw ? 24 : 16);
  int k = kt * 32 + halfw * 16 + e;
  float v;
  if (transHeads){                  // A[m=h*F+g, k=f] = W[h, f, g]
    int h = m / F, g = m % F;
    v = W[((size_t)h * F + k) * F + g];
  } else {                          // A[m,k] = W[m*Kd + k]
    v = W[(size_t)m * Kd + k];
  }
  out[p] = (_Float16)v;
}

// ---------------- sinusoidal PE of relative coords, column-major f16 -----------
__global__ void k_pe(const float* __restrict__ pos, const int* __restrict__ idx,
                     _Float16* __restrict__ X){
  int col = blockIdx.x * blockDim.x + threadIdx.x;
  if (col >= COLS) return;
  int bn = col >> 4, k = col & 15;
  int b = bn / NN, n = bn % NN;
  int nb = idx[(size_t)bn * KK + k];
  const float* pb = pos + (size_t)b * NN * 2;
  float r0 = pb[nb * 2]     - pb[n * 2];
  float r1 = pb[nb * 2 + 1] - pb[n * 2 + 1];
  _Float16* xc = X + (size_t)col * PE_DIM;
  const float ln1e4_over_nf = 9.210340371976184f / 16.f;
  for (int j = 0; j < 16; ++j){
    float f  = __expf(-(float)j * ln1e4_over_nf);
    float a0 = r0 * f, a1 = r1 * f;
    xc[j]        = (_Float16)__sinf(a0);
    xc[16 + j]   = (_Float16)__cosf(a0);
    xc[32 + j]   = (_Float16)__sinf(a1);
    xc[48 + j]   = (_Float16)__cosf(a1);
  }
}

// ---------------- WMMA GEMM: out[M x cols] = A[M x Kd] * X[Kd x cols] ----------
// X is column-major f16 (Kd contiguous per column). One wave -> 16(M) x 64(N).
// Software-pipelined: fragments for step kt+1 are fetched while step kt's four
// WMMAs execute, so waits become partial (loadcnt<=in-flight prefetch) instead
// of draining to zero before every matrix op.
// mode 0: BN+LeakyReLU -> f16 col-major ; 1: raw -> f16 col-major ; 2: BN+LReLU -> f32
__global__ __launch_bounds__(256) void k_gemm(
    const _Float16* __restrict__ A, const _Float16* __restrict__ Xb,
    int M, int Kd, int cols, int mode,
    const float* __restrict__ gam, const float* __restrict__ bet,
    _Float16* __restrict__ outH, float* __restrict__ outF){
  int wid  = blockIdx.x * 8 + (threadIdx.x >> 5);
  int lane = threadIdx.x & 31;
  int nM = M >> 4;
  int mt = wid % nM;
  int colbase = (wid / nM) * 64;
  if (colbase >= cols) return;
  int nK = Kd >> 5;
  int halfw = lane >> 4, ln = lane & 15;

  const v16h* Ap = (const v16h*)A + (size_t)mt * nK * 32 + lane;
  const _Float16* B0 = Xb + (size_t)(colbase + ln     ) * Kd + halfw * 16;
  const _Float16* B1 = Xb + (size_t)(colbase + ln + 16) * Kd + halfw * 16;
  const _Float16* B2 = Xb + (size_t)(colbase + ln + 32) * Kd + halfw * 16;
  const _Float16* B3 = Xb + (size_t)(colbase + ln + 48) * Kd + halfw * 16;

  v8f c0 = {}, c1 = {}, c2 = {}, c3 = {};

  // prologue: stage kt=0 fragments
  v16h a  = Ap[0];
  v16h b0 = *(const v16h*)(B0);
  v16h b1 = *(const v16h*)(B1);
  v16h b2 = *(const v16h*)(B2);
  v16h b3 = *(const v16h*)(B3);

  #pragma unroll 2
  for (int kt = 0; kt < nK - 1; ++kt){
    // prefetch next step's fragments before consuming current ones
    size_t o = (size_t)(kt + 1) * 32;
    v16h an  = Ap[o];
    v16h bn0 = *(const v16h*)(B0 + o);
    v16h bn1 = *(const v16h*)(B1 + o);
    v16h bn2 = *(const v16h*)(B2 + o);
    v16h bn3 = *(const v16h*)(B3 + o);
    c0 = __builtin_amdgcn_wmma_f32_16x16x32_f16(false, a, false, b0, (short)0, c0, false, false);
    c1 = __builtin_amdgcn_wmma_f32_16x16x32_f16(false, a, false, b1, (short)0, c1, false, false);
    c2 = __builtin_amdgcn_wmma_f32_16x16x32_f16(false, a, false, b2, (short)0, c2, false, false);
    c3 = __builtin_amdgcn_wmma_f32_16x16x32_f16(false, a, false, b3, (short)0, c3, false, false);
    a = an; b0 = bn0; b1 = bn1; b2 = bn2; b3 = bn3;
  }
  // epilogue step
  c0 = __builtin_amdgcn_wmma_f32_16x16x32_f16(false, a, false, b0, (short)0, c0, false, false);
  c1 = __builtin_amdgcn_wmma_f32_16x16x32_f16(false, a, false, b1, (short)0, c1, false, false);
  c2 = __builtin_amdgcn_wmma_f32_16x16x32_f16(false, a, false, b2, (short)0, c2, false, false);
  c3 = __builtin_amdgcn_wmma_f32_16x16x32_f16(false, a, false, b3, (short)0, c3, false, false);

  // C/D layout: lane -> col = lane%16 (+16j), rows = (lane/16)*8 + r
  int rowb = mt * 16 + halfw * 8;
  float gg[8], bb[8];
  if (mode != 1){
    #pragma unroll
    for (int r = 0; r < 8; ++r){ gg[r] = gam[rowb + r]; bb[r] = bet[rowb + r]; }
  }
  v8f cc[4] = {c0, c1, c2, c3};
  #pragma unroll
  for (int jj = 0; jj < 4; ++jj){
    int colc = colbase + jj * 16 + ln;
    #pragma unroll
    for (int r = 0; r < 8; ++r){
      float v = cc[jj][r];
      int row = rowb + r;
      if (mode == 1){
        outH[(size_t)colc * M + row] = (_Float16)v;
      } else {
        v = lrelu_(v * INVS * gg[r] + bb[r]);
        if (mode == 0) outH[(size_t)colc * M + row] = (_Float16)v;
        else           outF[(size_t)colc * M + row] = v;
      }
    }
  }
}

// ---------------- attention glue: coef gather, softmax over K, elu, sum -------
// Wh: f16 col-major [(b,n,k)][H*F]; writes XC[(b,n)][off + h*F+g]
__global__ void k_attn(const _Float16* __restrict__ Wh, const int* __restrict__ types,
                       const int* __restrict__ idx, const float* __restrict__ ap,
                       const float* __restrict__ as_, int F, int off,
                       _Float16* __restrict__ XC){
  long long total = (long long)BB * NN * HH * F;
  long long t = (long long)blockIdx.x * blockDim.x + threadIdx.x;
  if (t >= total) return;
  int M = HH * F;
  int bn = (int)(t / M);
  int m  = (int)(t % M);
  int h = m / F, g = m % F;
  int b = bn / NN;
  int core = types[bn];
  const _Float16* whp = Wh + (size_t)bn * KK * M + m;
  int idx0 = idx[(size_t)bn * KK];
  int t0 = types[b * NN + idx0];
  float wh[KK], ee[KK];
  float mx = -1e30f;
  for (int k = 0; k < KK; ++k){
    int nb = idx[(size_t)bn * KK + k];
    int tc = types[b * NN + nb];
    int s0 = tc < core ? tc : core;
    int s1 = tc < core ? core : tc;
    int pidx = s0 * CC - (s0 * (s0 - 1)) / 2 + (s1 - s0);
    float coef;
    if (k == 0 && t0 < core) coef = as_[((size_t)h * CC + core) * F + g];
    else                     coef = ap [((size_t)h * PPAIR + pidx) * F + g];
    float w = (float)whp[(size_t)k * M];
    wh[k] = w;
    float e = lrelu_(w * coef);
    ee[k] = e;
    mx = fmaxf(mx, e);
  }
  float s = 0.f;
  for (int k = 0; k < KK; ++k){ ee[k] = __expf(ee[k] - mx); s += ee[k]; }
  float inv = 1.f / s, acc = 0.f;
  for (int k = 0; k < KK; ++k){
    float o = ee[k] * inv * wh[k];
    acc += (o > 0.f) ? o : (__expf(o) - 1.f);   // elu
  }
  XC[(size_t)bn * XCF + off + m] = (_Float16)acc;
}

// ---------------- edge features: [neighbor - center, center], col-major f16 ----
__global__ void k_edge(const _Float16* __restrict__ XC, const int* __restrict__ idx,
                       int Fin, int off, _Float16* __restrict__ Xout){
  long long total = (long long)COLS * 2 * Fin;
  long long t = (long long)blockIdx.x * blockDim.x + threadIdx.x;
  if (t >= total) return;
  int F2 = 2 * Fin;
  int col = (int)(t / F2);
  int c   = (int)(t % F2);
  int bn = col >> 4, k = col & 15;
  int b = bn / NN;
  int f = (c < Fin) ? c : (c - Fin);
  float ctr = (float)XC[(size_t)bn * XCF + off + f];
  float v;
  if (c < Fin){
    int nb = idx[(size_t)bn * KK + k];
    v = (float)XC[((size_t)(b * NN + nb)) * XCF + off + f] - ctr;
  } else {
    v = ctr;
  }
  Xout[(size_t)col * F2 + c] = (_Float16)v;
}

// ---------------- global max+mean pool over N --------------------------------
__global__ void k_pool(const float* __restrict__ Y5, float* __restrict__ feat){
  int t = blockIdx.x * blockDim.x + threadIdx.x;
  if (t >= BB * EMBD) return;
  int b = t / EMBD, c = t % EMBD;
  float mx = -1e30f, s = 0.f;
  for (int n = 0; n < NN; ++n){
    float v = Y5[((size_t)(b * NN + n)) * EMBD + c];
    mx = fmaxf(mx, v); s += v;
  }
  feat[(size_t)b * 2 * EMBD + c]        = mx;
  feat[(size_t)b * 2 * EMBD + EMBD + c] = s / (float)NN;
}

// ---------------- tiny head MLP (one block per batch) ------------------------
__global__ __launch_bounds__(512) void k_mlp(
    const float* __restrict__ feat, const float* __restrict__ l1w,
    const float* __restrict__ g6, const float* __restrict__ b6,
    const float* __restrict__ l2w, const float* __restrict__ l2b,
    const float* __restrict__ g7, const float* __restrict__ b7,
    const float* __restrict__ l3w, const float* __restrict__ l3b,
    float* __restrict__ out){
  __shared__ float sf[2 * EMBD];
  __shared__ float h1[512];
  __shared__ float h2[256];
  int b = blockIdx.x, tid = threadIdx.x;
  for (int i = tid; i < 2 * EMBD; i += 512) sf[i] = feat[(size_t)b * 2 * EMBD + i];
  __syncthreads();
  {
    float s = 0.f;
    const float* w = l1w + (size_t)tid * 2 * EMBD;
    for (int c = 0; c < 2 * EMBD; ++c) s += sf[c] * w[c];
    h1[tid] = lrelu_(s * INVS * g6[tid] + b6[tid]);
  }
  __syncthreads();
  if (tid < 256){
    float s = l2b[tid];
    const float* w = l2w + (size_t)tid * 512;
    for (int c = 0; c < 512; ++c) s += h1[c] * w[c];
    h2[tid] = lrelu_(s * INVS * g7[tid] + b7[tid]);
  }
  __syncthreads();
  if (tid < 2){
    float s = l3b[tid];
    const float* w = l3w + (size_t)tid * 256;
    for (int c = 0; c < 256; ++c) s += h2[c] * w[c];
    out[b * 2 + tid] = s;
  }
}

// =============================== launcher =====================================
extern "C" void kernel_launch(void* const* d_in, const int* in_sizes, int n_in,
                              void* d_out, int out_size, void* d_ws, size_t ws_size,
                              hipStream_t stream) {
  const float* pos    = (const float*)d_in[0];
  const int*   types  = (const int*)  d_in[1];
  const float* conv1w = (const float*)d_in[2];
  const float* bn1g = (const float*)d_in[3];  const float* bn1b = (const float*)d_in[4];
  const float* W1  = (const float*)d_in[5];  const float* ap1 = (const float*)d_in[6];  const float* as1 = (const float*)d_in[7];
  const float* conv2w = (const float*)d_in[8];
  const float* bn2g = (const float*)d_in[9];  const float* bn2b = (const float*)d_in[10];
  const float* W2  = (const float*)d_in[11]; const float* ap2 = (const float*)d_in[12]; const float* as2 = (const float*)d_in[13];
  const float* conv3w = (const float*)d_in[14];
  const float* bn3g = (const float*)d_in[15]; const float* bn3b = (const float*)d_in[16];
  const float* W3  = (const float*)d_in[17]; const float* ap3 = (const float*)d_in[18]; const float* as3 = (const float*)d_in[19];
  const float* conv4w = (const float*)d_in[20];
  const float* bn4g = (const float*)d_in[21]; const float* bn4b = (const float*)d_in[22];
  const float* W4  = (const float*)d_in[23]; const float* ap4 = (const float*)d_in[24]; const float* as4 = (const float*)d_in[25];
  const float* conv5w = (const float*)d_in[26];
  const float* bn5g = (const float*)d_in[27]; const float* bn5b = (const float*)d_in[28];
  const float* lin1w = (const float*)d_in[29];
  const float* bn6g = (const float*)d_in[30]; const float* bn6b = (const float*)d_in[31];
  const float* lin2w = (const float*)d_in[32]; const float* lin2b = (const float*)d_in[33];
  const float* bn7g = (const float*)d_in[34]; const float* bn7b = (const float*)d_in[35];
  const float* lin3w = (const float*)d_in[36]; const float* lin3b = (const float*)d_in[37];
  float* out = (float*)d_out;

  // -------- workspace carve (peak ~200 MB; Xbuf reused for edge-in / Wh / Y5) --
  char* ws = (char*)d_ws;
  size_t cur = 0;
  auto carve = [&](size_t bytes) -> char* {
    cur = (cur + 255) & ~(size_t)255;
    char* p = ws + cur; cur += bytes; return p;
  };
  int*      idxb = (int*)     carve((size_t)BB * NN * KK * 4);
  _Float16* pc1  = (_Float16*)carve((size_t)64 * 64 * 2);
  _Float16* pc2  = (_Float16*)carve((size_t)64 * 512 * 2);
  _Float16* pc3  = (_Float16*)carve((size_t)128 * 512 * 2);
  _Float16* pc4  = (_Float16*)carve((size_t)256 * 1024 * 2);
  _Float16* pc5  = (_Float16*)carve((size_t)1024 * 2048 * 2);
  _Float16* pw1  = (_Float16*)carve((size_t)256 * 64 * 2);
  _Float16* pw2  = (_Float16*)carve((size_t)256 * 64 * 2);
  _Float16* pw3  = (_Float16*)carve((size_t)512 * 128 * 2);
  _Float16* pw4  = (_Float16*)carve((size_t)1024 * 256 * 2);
  _Float16* Xbuf = (_Float16*)carve((size_t)COLS * 1024 * 2);   // 128 MB
  _Float16* Ybuf = (_Float16*)carve((size_t)COLS * 256 * 2);    // 32 MB
  _Float16* XC   = (_Float16*)carve((size_t)PCOLS * XCF * 2);   // 16 MB
  float*    feat = (float*)   carve((size_t)BB * 2 * EMBD * 4);
  float*    Y5   = (float*)Xbuf;                                // reuse

  auto cdiv = [](long long a, long long b) -> int { return (int)((a + b - 1) / b); };

  k_ball<<<cdiv(BB * NN, 256), 256, 0, stream>>>(pos, idxb);

  auto pack = [&](const float* W, _Float16* o, int M, int Kd, int F, int tr){
    k_pack<<<cdiv((long long)M * Kd, 256), 256, 0, stream>>>(W, o, M, Kd, F, tr);
  };
  pack(conv1w, pc1,   64,   64,   0, 0);
  pack(conv2w, pc2,   64,  512,   0, 0);
  pack(conv3w, pc3,  128,  512,   0, 0);
  pack(conv4w, pc4,  256, 1024,   0, 0);
  pack(conv5w, pc5, 1024, 2048,   0, 0);
  pack(W1, pw1,  256,  64,  64, 1);
  pack(W2, pw2,  256,  64,  64, 1);
  pack(W3, pw3,  512, 128, 128, 1);
  pack(W4, pw4, 1024, 256, 256, 1);

  auto gemm = [&](const _Float16* A, const _Float16* Xin, int M, int Kd, int cols,
                  int mode, const float* g, const float* bv,
                  _Float16* oh, float* of){
    int waves = (M / 16) * (cols / 64);
    k_gemm<<<waves / 8, 256, 0, stream>>>(A, Xin, M, Kd, cols, mode, g, bv, oh, of);
  };

  // layer 1: PE -> conv1 -> Wh1 -> attn1 (x1 at XC[0:256))
  k_pe<<<COLS / 256, 256, 0, stream>>>(pos, idxb, Xbuf);
  gemm(pc1, Xbuf,  64,  64, COLS, 0, bn1g, bn1b, Ybuf, nullptr);
  gemm(pw1, Ybuf, 256,  64, COLS, 1, nullptr, nullptr, Xbuf, nullptr);
  k_attn<<<cdiv((long long)BB * NN * HH * 64, 256), 256, 0, stream>>>(
      Xbuf, types, idxb, ap1, as1, 64, 0, XC);

  // layer 2: edge(x1) -> conv2 -> Wh2 -> attn2 (x2 at XC[256:512))
  k_edge<<<cdiv((long long)COLS * 512, 256), 256, 0, stream>>>(XC, idxb, 256, 0, Xbuf);
  gemm(pc2, Xbuf,  64, 512, COLS, 0, bn2g, bn2b, Ybuf, nullptr);
  gemm(pw2, Ybuf, 256,  64, COLS, 1, nullptr, nullptr, Xbuf, nullptr);
  k_attn<<<cdiv((long long)BB * NN * HH * 64, 256), 256, 0, stream>>>(
      Xbuf, types, idxb, ap2, as2, 64, 256, XC);

  // layer 3: edge(x2) -> conv3 -> Wh3 -> attn3 (x3 at XC[512:1024))
  k_edge<<<cdiv((long long)COLS * 512, 256), 256, 0, stream>>>(XC, idxb, 256, 256, Xbuf);
  gemm(pc3, Xbuf, 128, 512, COLS, 0, bn3g, bn3b, Ybuf, nullptr);
  gemm(pw3, Ybuf, 512, 128, COLS, 1, nullptr, nullptr, Xbuf, nullptr);
  k_attn<<<cdiv((long long)BB * NN * HH * 128, 256), 256, 0, stream>>>(
      Xbuf, types, idxb, ap3, as3, 128, 512, XC);

  // layer 4: edge(x3) -> conv4 -> Wh4 -> attn4 (x4 at XC[1024:2048))
  k_edge<<<cdiv((long long)COLS * 1024, 256), 256, 0, stream>>>(XC, idxb, 512, 512, Xbuf);
  gemm(pc4, Xbuf,  256, 1024, COLS, 0, bn4g, bn4b, Ybuf, nullptr);
  gemm(pw4, Ybuf, 1024,  256, COLS, 1, nullptr, nullptr, Xbuf, nullptr);
  k_attn<<<cdiv((long long)BB * NN * HH * 256, 256), 256, 0, stream>>>(
      Xbuf, types, idxb, ap4, as4, 256, 1024, XC);

  // conv5 on concat (per-point cols) -> f32, then pool + MLP head
  gemm(pc5, XC, 1024, 2048, PCOLS, 2, bn5g, bn5b, nullptr, Y5);
  k_pool<<<cdiv(BB * EMBD, 256), 256, 0, stream>>>(Y5, feat);
  k_mlp<<<BB, 512, 0, stream>>>(feat, lin1w, bn6g, bn6b, lin2w, lin2b,
                                bn7g, bn7b, lin3w, lin3b, out);
}
